// _Expert_2834678415365
// MI455X (gfx1250) — matrix operand order repaired and checked by reading, not compiled, not run on hardware
//
#include <hip/hip_runtime.h>
#include <hip/hip_bf16.h>

// ---------------------------------------------------------------------------
// MoE FFN: grouped GEMM (bf16 WMMA, fp32 accum) -> erf-GELU -> grouped GEMM.
// Tokens are contiguous per expert; per-expert segments come from a device
// prefix sum of fwd_expert_count. Row tiles are launched per expert with
// uniform early exit, loads clamped to valid memory, stores masked at the
// segment boundary (EXEC stays all-ones at every WMMA).
// ---------------------------------------------------------------------------

typedef __attribute__((ext_vector_type(16))) __bf16 v16bf;
typedef __attribute__((ext_vector_type(8)))  float  v8f;
typedef __attribute__((ext_vector_type(4)))  int    v4i_t;

// Address-space-qualified element types for the async-load builtin
// (diagnostic showed arg0 = "int __vector_size(16) __device__ *", i.e. AS1 int4*).
typedef __attribute__((address_space(1))) v4i_t as1_v4i;
typedef __attribute__((address_space(3))) v4i_t as3_v4i;

#define TILE_M 128
#define TILE_N 128
#define TILE_K 32
#define LDS_STRIDE 40   // 32 halves + 8 pad halves (80B row stride)

#if defined(__has_builtin)
#if __has_builtin(__builtin_amdgcn_global_load_async_to_lds_b128) && \
    __has_builtin(__builtin_amdgcn_s_wait_asynccnt)
#define USE_ASYNC_LDS 1
#endif
#endif

__device__ __forceinline__ unsigned short f2bf_rne(float x) {
  unsigned u = __float_as_uint(x);
  u += 0x7FFFu + ((u >> 16) & 1u);
  return (unsigned short)(u >> 16);
}

__device__ __forceinline__ float gelu_erf(float x) {
  return 0.5f * x * (1.0f + erff(x * 0.70710678118654752f));
}

struct Frag32B { uint4 lo; uint4 hi; };

__device__ __forceinline__ void stage16(const unsigned short* g, unsigned short* l) {
#ifdef USE_ASYNC_LDS
  __builtin_amdgcn_global_load_async_to_lds_b128(
      (as1_v4i*)g, (as3_v4i*)l, 0, 0);
#else
  *(uint4*)l = *(const uint4*)g;
#endif
}

// -------------------------- preprocessing kernels --------------------------

__global__ __launch_bounds__(256) void f32_to_bf16_kernel(
    const float* __restrict__ s, unsigned short* __restrict__ d, long n) {
  long i = ((long)blockIdx.x * blockDim.x + threadIdx.x) * 4;
  if (i >= n) return;
  float4 f = *(const float4*)(s + i);
  unsigned long long p =  (unsigned long long)f2bf_rne(f.x)
                       | ((unsigned long long)f2bf_rne(f.y) << 16)
                       | ((unsigned long long)f2bf_rne(f.z) << 32)
                       | ((unsigned long long)f2bf_rne(f.w) << 48);
  *(unsigned long long*)(d + i) = p;
}

__global__ void csum_kernel(const int* __restrict__ cnt, int* __restrict__ starts, int E) {
  if (threadIdx.x == 0 && blockIdx.x == 0) {
    int s = 0;
    starts[0] = 0;
    for (int e = 0; e < E; ++e) { s += cnt[e]; starts[e + 1] = s; }
  }
}

// ------------------------------ grouped GEMM -------------------------------
// A: [T, K] bf16 row-major.  W: [E, N, K] bf16 (K contiguous => W row n is a
// ready-made column-major B fragment source).  out = A @ W[e]^T + bias[e].

template <bool GELU, bool OUT_BF16>
__global__ __launch_bounds__(256) void grouped_gemm_bf16(
    const unsigned short* __restrict__ A,
    const unsigned short* __restrict__ W,
    const float* __restrict__ bias,
    void* __restrict__ outv,
    const int* __restrict__ starts,
    int T, int N, int K) {
  __shared__ __align__(16) unsigned short ldsA[TILE_M * LDS_STRIDE];
  __shared__ __align__(16) unsigned short ldsB[TILE_N * LDS_STRIDE];

  const int e    = blockIdx.z;
  const int seg1 = starts[e + 1];
  const int row0 = starts[e] + blockIdx.x * TILE_M;
  if (row0 >= seg1) return;                 // uniform across the block
  const int tileN = blockIdx.y * TILE_N;

  const int tid  = threadIdx.x;
  const int lane = tid & 31;
  const int wave = tid >> 5;
  const int wm   = wave & 1;                // 64-row quadrant
  const int wn   = wave >> 1;               // 32-col slice
  const int half = lane >> 4;               // lane group 0/1
  const int l16  = lane & 15;

  v8f acc[4][2];
  const v8f vzero = {0.f, 0.f, 0.f, 0.f, 0.f, 0.f, 0.f, 0.f};
  for (int am = 0; am < 4; ++am)
    for (int bn = 0; bn < 2; ++bn) acc[am][bn] = vzero;

  const unsigned short* Wb = W + (size_t)e * N * K;

  for (int kt = 0; kt < K; kt += TILE_K) {
    // Stage 128x32 A tile and 128x32 B tile: 512 16B chunks each,
    // 2 chunks per thread per tile.
    for (int i = 0; i < 2; ++i) {
      int c  = tid + i * 256;       // 0..511
      int r  = c >> 2;              // 0..127
      int kc = (c & 3) * 8;         // half offset 0/8/16/24
      int gr = row0 + r;
      if (gr >= T) gr = T - 1;      // clamp: junk rows are masked at store
      stage16(A + (size_t)gr * K + kt + kc, &ldsA[r * LDS_STRIDE + kc]);
      stage16(Wb + (size_t)(tileN + r) * K + kt + kc, &ldsB[r * LDS_STRIDE + kc]);
    }
#ifdef USE_ASYNC_LDS
    __builtin_amdgcn_s_wait_asynccnt(0);
#endif
    __syncthreads();

    // B fragments: lane l -> col = l&15, K = 8*(l>>4)+{0..7} and +16.
    v16bf bfrag[2];
    for (int bn = 0; bn < 2; ++bn) {
      const unsigned short* bp =
          &ldsB[(wn * 32 + bn * 16 + l16) * LDS_STRIDE + half * 8];
      Frag32B f;
      f.lo = *(const uint4*)bp;
      f.hi = *(const uint4*)(bp + 16);
      bfrag[bn] = __builtin_bit_cast(v16bf, f);
    }
    // A fragments: lane l -> row = l&15, K = 8*(l>>4)+{0..7} and +16.
    for (int am = 0; am < 4; ++am) {
      const unsigned short* ap =
          &ldsA[(wm * 64 + am * 16 + l16) * LDS_STRIDE + half * 8];
      Frag32B f;
      f.lo = *(const uint4*)ap;
      f.hi = *(const uint4*)(ap + 16);
      v16bf afrag = __builtin_bit_cast(v16bf, f);
      for (int bn = 0; bn < 2; ++bn) {
        acc[am][bn] = __builtin_amdgcn_wmma_f32_16x16x32_bf16(
            false, afrag, false, bfrag[bn], (short)0, acc[am][bn], false, false);
      }
    }
    __syncthreads();
  }

  // Epilogue: C/D layout -> lane l holds col = l&15, rows 8*(l>>4)+j in VGPR j.
  for (int bn = 0; bn < 2; ++bn) {
    int col  = tileN + wn * 32 + bn * 16 + l16;
    float bb = bias[(size_t)e * N + col];
    for (int am = 0; am < 4; ++am) {
      int rbase = row0 + wm * 64 + am * 16 + half * 8;
      for (int j = 0; j < 8; ++j) {
        int r = rbase + j;
        if (r < seg1) {
          float v = acc[am][bn][j] + bb;
          if (GELU) v = gelu_erf(v);
          if (OUT_BF16)
            ((unsigned short*)outv)[(size_t)r * N + col] = f2bf_rne(v);
          else
            ((float*)outv)[(size_t)r * N + col] = v;
        }
      }
    }
  }
}

// ------------------------------- launch ------------------------------------

static inline size_t alignUp256(size_t x) { return (x + 255) & ~(size_t)255; }

extern "C" void kernel_launch(void* const* d_in, const int* in_sizes, int n_in,
                              void* d_out, int out_size, void* d_ws, size_t ws_size,
                              hipStream_t stream) {
  const float* inp = (const float*)d_in[0];
  const float* w1  = (const float*)d_in[1];
  const float* b1  = (const float*)d_in[2];
  const float* w2  = (const float*)d_in[3];
  const float* b2  = (const float*)d_in[4];
  const int*   cnt = (const int*)d_in[5];

  const long nInp = in_sizes[0];
  const long nW1  = in_sizes[1];
  const long nW2  = in_sizes[3];
  const int  E    = in_sizes[5];
  const int  H    = in_sizes[2] / E;
  const int  D    = in_sizes[4] / E;
  const int  T    = (int)(nInp / D);

  // Workspace carve: bf16 copies of inp/w1/w2, bf16 hidden, expert offsets.
  char* p = (char*)d_ws;
  unsigned short* inpB = (unsigned short*)p; p += alignUp256((size_t)nInp * 2);
  unsigned short* w1B  = (unsigned short*)p; p += alignUp256((size_t)nW1 * 2);
  unsigned short* w2B  = (unsigned short*)p; p += alignUp256((size_t)nW2 * 2);
  unsigned short* hB   = (unsigned short*)p; p += alignUp256((size_t)T * H * 2);
  int* starts          = (int*)p;

  const int thr = 256;
  f32_to_bf16_kernel<<<(int)((nInp / 4 + thr - 1) / thr), thr, 0, stream>>>(inp, inpB, nInp);
  f32_to_bf16_kernel<<<(int)((nW1  / 4 + thr - 1) / thr), thr, 0, stream>>>(w1,  w1B,  nW1);
  f32_to_bf16_kernel<<<(int)((nW2  / 4 + thr - 1) / thr), thr, 0, stream>>>(w2,  w2B,  nW2);
  csum_kernel<<<1, 32, 0, stream>>>(cnt, starts, E);

  dim3 g1((T + TILE_M - 1) / TILE_M, H / TILE_N, E);
  grouped_gemm_bf16<true, true><<<g1, 256, 0, stream>>>(inpB, w1B, b1, hB, starts, T, H, D);

  dim3 g2((T + TILE_M - 1) / TILE_M, D / TILE_N, E);
  grouped_gemm_bf16<false, false><<<g2, 256, 0, stream>>>(hB, w2B, b2, d_out, starts, T, D, H);
}
